// NeuralSheet_64922725646795
// MI455X (gfx1250) — compile-verified, matrix operand order checked by reading, not run on hardware
//
#include <hip/hip_runtime.h>
#include <hip/hip_bf16.h>
#include <math.h>

#define S2 4096
#define NITER 50

typedef __attribute__((ext_vector_type(16))) __bf16 v16bf;
typedef __attribute__((ext_vector_type(8)))  float  v8f;

union PackU { uint4 u[2]; v16bf v; };

static __device__ __forceinline__ v16bf pack16(uint4 lo, uint4 hi) {
    PackU t; t.u[0] = lo; t.u[1] = hi; return t.v;
}

// float -> bf16 with round-to-nearest-even
static __device__ __forceinline__ unsigned short f2bf(float f) {
    unsigned u = __float_as_uint(f);
    u += 0x7FFFu + ((u >> 16) & 1u);
    return (unsigned short)(u >> 16);
}

// Fragment-swizzled index for element (row, col) of a 4096x4096 bf16 matrix.
// Layout: [strip = row/16][chunk = col/32][lane][elem], 16 bf16 (32B) per lane,
// matching the v_wmma_f32_16x16x32_bf16 A-operand per-lane layout:
//   lane = (row%16) + 16*hi   where hi = (col%32 in [8..15]|[24..31])
//   elem = (col%8) + 8*((col%32) >= 16)
// so a wave's (strip, chunk) fragment is one contiguous 1KB block.
static __device__ __forceinline__ size_t swz_idx(int row, int col) {
    int t     = row >> 4;
    int m     = row & 15;
    int chunk = col >> 5;
    int kk    = col & 31;
    int hi    = (kk >> 3) & 1;          // K in [8..15] or [24..31] -> upper lanes
    int lane  = m + (hi << 4);
    int elem  = (kk & 7) + ((kk & 16) >> 1);
    return ((((size_t)t * 128 + chunk) * 32) + lane) * 16 + elem;
}

// ---------------------------------------------------------------------------
// Precompute the two loop-invariant row-normalized matrices in bf16, stored
// in WMMA-fragment-swizzled order:
//   l4w  = lateral_weights_exc - normalize_row(l4_corr * (1 - masks))
//   latw = normalize_row(relu(lc - 1.5/S2)) - normalize_row(relu(lc - 1.0/S2))
// One row per workgroup; non-temporal reads keep the 256MB of f32 inputs from
// thrashing L2 (we want the 64MB of bf16 outputs resident for 50 iterations).
// ---------------------------------------------------------------------------
__global__ __launch_bounds__(256) void prep_mats(
    const float* __restrict__ lc,   const float* __restrict__ l4c,
    const float* __restrict__ lwe,  const float* __restrict__ masks,
    unsigned short* __restrict__ l4w, unsigned short* __restrict__ latw)
{
    const int row = blockIdx.x;
    const size_t base = (size_t)row * S2;
    const float cE = 1.5f / (float)S2;
    const float cI = 1.0f / (float)S2;

    float sm = 0.f, se = 0.f, si = 0.f;
    for (int c = threadIdx.x; c < S2; c += 256) {
        float l4cv = __builtin_nontemporal_load(l4c + base + c);
        float mk   = __builtin_nontemporal_load(masks + base + c);
        float lcv  = __builtin_nontemporal_load(lc + base + c);
        sm += l4cv * (1.0f - mk);
        se += fmaxf(lcv - cE, 0.0f);
        si += fmaxf(lcv - cI, 0.0f);
    }
    for (int off = 16; off; off >>= 1) {
        sm += __shfl_xor(sm, off, 32);
        se += __shfl_xor(se, off, 32);
        si += __shfl_xor(si, off, 32);
    }
    __shared__ float red[3][8];
    int wave = threadIdx.x >> 5, lane = threadIdx.x & 31;
    if (lane == 0) { red[0][wave] = sm; red[1][wave] = se; red[2][wave] = si; }
    __syncthreads();
    sm = 0.f; se = 0.f; si = 0.f;
    #pragma unroll
    for (int w = 0; w < 8; ++w) { sm += red[0][w]; se += red[1][w]; si += red[2][w]; }

    const float rm = 1.0f / (sm + 1e-11f);
    const float re = 1.0f / (se + 1e-11f);
    const float ri = 1.0f / (si + 1e-11f);

    for (int c = threadIdx.x; c < S2; c += 256) {
        float l4cv = __builtin_nontemporal_load(l4c + base + c);
        float mk   = __builtin_nontemporal_load(masks + base + c);
        float lcv  = __builtin_nontemporal_load(lc + base + c);
        float lwev = __builtin_nontemporal_load(lwe + base + c);
        float l4wv = lwev - (l4cv * (1.0f - mk)) * rm;
        float latv = fmaxf(lcv - cE, 0.0f) * re - fmaxf(lcv - cI, 0.0f) * ri;
        size_t di = swz_idx(row, c);
        l4w[di]  = f2bf(l4wv);
        latw[di] = f2bf(latv);
    }
}

// ---------------------------------------------------------------------------
// afferent[u] = sum_{t<225} img[rf[u,t,0], rf[u,t,1]] * aff_w[u,t]
// one wave32 per sheet unit (8 units / block)
// ---------------------------------------------------------------------------
__global__ __launch_bounds__(256) void aff_kernel(
    const float* __restrict__ img, const int* __restrict__ rf,
    const float* __restrict__ aw,  float* __restrict__ aff)
{
    int wave = threadIdx.x >> 5, lane = threadIdx.x & 31;
    int u = blockIdx.x * 8 + wave;
    const int*   g = rf + (size_t)u * 450;
    const float* w = aw + (size_t)u * 225;
    float s = 0.f;
    for (int t = lane; t < 225; t += 32) {
        int r = g[2 * t];
        int c = g[2 * t + 1];
        s += img[r * 48 + c] * w[t];
    }
    for (int off = 16; off; off >>= 1) s += __shfl_xor(s, off, 32);
    if (lane == 0) aff[u] = s;
}

__global__ __launch_bounds__(256) void zero_vec(float* __restrict__ a, float* __restrict__ b) {
    int i = blockIdx.x * 256 + threadIdx.x;
    if (i < S2) { a[i] = 0.0f; b[i] = 0.0f; }
}

__global__ __launch_bounds__(256) void copy_out(const float* __restrict__ src, float* __restrict__ dst) {
    int i = blockIdx.x * 256 + threadIdx.x;
    if (i < S2) dst[i] = src[i];
}

// ---------------------------------------------------------------------------
// One recurrence step (grid 256 x 256 threads, 16 output rows per WG):
//   y1 = l4w @ l4 ; y2 = latw @ cur        (bf16 WMMA GEMV, L2 resident)
//   l4'  = tanh(relu(aff*0.5 + cur*(b*0.5) + y1 - l4thr) * 2)
//   cur' = tanh(relu(l4' + y2 - thr))
// Wave mapping: part = wave>>2 (0: l4w@l4, 1: latw@cur), kq = wave&3 is the
// K-quarter (1024 of K=4096). A fragments stream from the swizzled layout as
// contiguous 1KB-per-wave-chunk blocks; B = vector replicated across all 16
// N-columns, so lane 0 / lane 16 accumulators hold rows 0..7 / 8..15.
// ---------------------------------------------------------------------------
__global__ __launch_bounds__(256) void iter_kernel(
    const unsigned short* __restrict__ l4w,
    const unsigned short* __restrict__ latw,
    const float* __restrict__ aff,
    const float* __restrict__ thr,
    const float* __restrict__ l4thr,
    const float* __restrict__ curIn,
    const float* __restrict__ l4In,
    float* __restrict__ curOut,
    float* __restrict__ l4Out)
{
    __shared__ unsigned short s_l4[S2];
    __shared__ unsigned short s_cur[S2];
    __shared__ float s_res[2][4][16];

    const int tid = threadIdx.x;
    for (int i = tid; i < S2; i += 256) {
        s_l4[i]  = f2bf(l4In[i]);
        s_cur[i] = f2bf(curIn[i]);
    }
    __syncthreads();

    const int wave = tid >> 5;
    const int lane = tid & 31;
    const int part = wave >> 2;        // 0: l4w@l4, 1: latw@cur
    const int kq   = wave & 3;         // K quarter: chunks [kq*32, kq*32+32)

    const unsigned short* M = part ? latw : l4w;
    const unsigned short* V = part ? s_cur : s_l4;

    // swizzled A: strip = blockIdx.x, fragment block = 512 ushorts per chunk
    const unsigned short* abase = M + (size_t)blockIdx.x * 128 * 512 + lane * 16;
    const int g = (lane & 16) >> 1;    // 0 for lanes 0-15, 8 for lanes 16-31
    const unsigned short* vb = V + g;

    v8f c = {0.f, 0.f, 0.f, 0.f, 0.f, 0.f, 0.f, 0.f};

    const int c0 = kq * 32;
    #pragma unroll 8
    for (int ch = c0; ch < c0 + 32; ++ch) {
        const unsigned short* ap = abase + (size_t)ch * 512;
        uint4 alo = *(const uint4*)(ap);        // lane's 16 bf16 fragment
        uint4 ahi = *(const uint4*)(ap + 8);
        int k = ch << 5;
        uint4 blo = *(const uint4*)(vb + k);    // vector chunk (broadcast)
        uint4 bhi = *(const uint4*)(vb + k + 16);
        v16bf a = pack16(alo, ahi);
        v16bf b = pack16(blo, bhi);
        c = __builtin_amdgcn_wmma_f32_16x16x32_bf16(false, a, false, b,
                                                    (short)0, c, false, false);
    }

    // D 16x16 f32: lane<16 vgpr j -> D[j][lane]; lane>=16 -> D[j+8][lane-16].
    // All columns identical => lane 0 holds rows 0..7, lane 16 rows 8..15.
    if (lane == 0) {
        #pragma unroll
        for (int j = 0; j < 8; ++j) s_res[part][kq][j] = c[j];
    } else if (lane == 16) {
        #pragma unroll
        for (int j = 0; j < 8; ++j) s_res[part][kq][8 + j] = c[j];
    }
    __syncthreads();

    if (tid < 16) {
        const int gr = blockIdx.x * 16 + tid;
        float y1 = s_res[0][0][tid] + s_res[0][1][tid] + s_res[0][2][tid] + s_res[0][3][tid];
        float y2 = s_res[1][0][tid] + s_res[1][1][tid] + s_res[1][2][tid] + s_res[1][3][tid];
        const float BH = (float)(0.5 / (0.5 + 1e-11) * 0.5);  // b * 0.5
        float l4_aff  = aff[gr] * 0.5f + curIn[gr] * BH;
        float l4_new  = tanhf(fmaxf(l4_aff + y1 - l4thr[gr], 0.0f) * 2.0f);
        float cur_new = tanhf(fmaxf(l4_new + y2 - thr[gr], 0.0f));
        l4Out[gr]  = l4_new;
        curOut[gr] = cur_new;
    }
}

extern "C" void kernel_launch(void* const* d_in, const int* in_sizes, int n_in,
                              void* d_out, int out_size, void* d_ws, size_t ws_size,
                              hipStream_t stream)
{
    const float* input_crop = (const float*)d_in[0];   // (1,1,48,48) f32
    const int*   rf_grids   = (const int*)  d_in[1];   // (4096,15,15,2) i32
    const float* aff_w      = (const float*)d_in[2];   // (4096,1,15,15) f32
    const float* lc         = (const float*)d_in[3];   // (4096,4096) f32
    const float* l4c        = (const float*)d_in[4];   // (4096,4096) f32
    const float* lwe        = (const float*)d_in[5];   // (4096,4096) f32
    const float* masks      = (const float*)d_in[6];   // (4096,4096) f32
    const float* thr        = (const float*)d_in[7];   // (1,1,64,64) f32
    const float* l4thr      = (const float*)d_in[8];   // (1,1,64,64) f32

    // workspace: two swizzled bf16 matrices (64MB) + afferent + 4 state vectors
    unsigned short* l4w  = (unsigned short*)d_ws;
    unsigned short* latw = l4w + (size_t)S2 * S2;
    float* aff  = (float*)(latw + (size_t)S2 * S2);
    float* cur0 = aff  + S2;
    float* l40  = cur0 + S2;
    float* cur1 = l40  + S2;
    float* l41  = cur1 + S2;

    prep_mats<<<S2, 256, 0, stream>>>(lc, l4c, lwe, masks, l4w, latw);
    aff_kernel<<<S2 / 8, 256, 0, stream>>>(input_crop, rf_grids, aff_w, aff);
    zero_vec<<<(S2 + 255) / 256, 256, 0, stream>>>(cur0, l40);

    for (int it = 0; it < NITER; ++it) {
        const float* ci = (it & 1) ? cur1 : cur0;
        const float* li = (it & 1) ? l41  : l40;
        float* co = (it & 1) ? cur0 : cur1;
        float* lo = (it & 1) ? l40  : l41;
        iter_kernel<<<256, 256, 0, stream>>>(l4w, latw, aff, thr, l4thr,
                                             ci, li, co, lo);
    }
    // it = 49 (odd) wrote into buffer 0 -> final cur lives in cur0
    copy_out<<<(S2 + 255) / 256, 256, 0, stream>>>(cur0, (float*)d_out);
}
